// FormulaNet_16157666968265
// MI455X (gfx1250) — matrix operand order repaired and checked by reading, not compiled, not run on hardware
//
#include <hip/hip_runtime.h>

#define LENGTH  50
#define OFFSETC 5
#define ENC_DIM 55
#define LSEQ    256
#define TTREES  2   // trees (batch rows) per workgroup

typedef __attribute__((ext_vector_type(16))) _Float16 v16h;
typedef __attribute__((ext_vector_type(8)))  _Float16 v8h;
typedef __attribute__((ext_vector_type(4)))  _Float16 v4h;
typedef __attribute__((ext_vector_type(8)))  float    v8f;

__global__ __launch_bounds__(256) void formula_net_kernel(
    const float* __restrict__ leaf_table, const float* __restrict__ op_table,
    const float* __restrict__ W_leaf,     const float* __restrict__ b_leaf,
    const float* __restrict__ W_bin,      const float* __restrict__ b_bin,
    const float* __restrict__ W_final,    const float* __restrict__ b_final,
    const int*   __restrict__ leaf_ids,   const int*   __restrict__ op_ids,
    float* __restrict__ out)
{
  __shared__ __align__(32) _Float16 sA[512 * 64];            // 64 KB
  __shared__ __align__(32) _Float16 sB[256 * 64];            // 32 KB (also embed staging)
  __shared__ __align__(32) _Float16 sOp[512 * 64];           // 64 KB (510 used + 2 pad rows)
  __shared__ v16h  sPackBin[24 * 32];                        // 24 KB
  __shared__ v16h  sPackLeaf[8 * 32];                        //  8 KB
  __shared__ float sBiasBin[64];
  __shared__ float sBiasLeaf[64];
  __shared__ float sRoot[TTREES * 64];

  const int tid    = threadIdx.x;
  const int lane   = tid & 31;
  const int wave   = tid >> 5;
  const int laneHi = lane >> 4;
  const int n15    = lane & 15;
  const int b0     = blockIdx.x * TTREES;
  const int ntw    = wave & 3;      // this wave's fixed N-tile

  if (tid < 32) {
    __builtin_prefetch((const char*)(leaf_ids + b0 * LSEQ) + tid * 64, 0, 0);
    __builtin_prefetch((const char*)(op_ids + b0 * 255) + tid * 64, 0, 0);
  }

  // ---------------- Stage 0: pack B operands + biases into LDS ----------------
  {
    _Float16* pb = (_Float16*)sPackBin;
    for (int idx = tid; idx < 24 * 32 * 16; idx += 256) {
      int j  = idx & 15;
      int ln = (idx >> 4) & 31;
      int f  = idx >> 9;
      int nt = f & 3, kc = f >> 2;
      int n    = nt * 16 + (ln & 15);
      int kloc = ((ln >> 4) << 4) + j;
      int seg  = kc >> 1;
      int ks   = ((kc & 1) << 5) + kloc;
      float v = 0.f;
      if (n < LENGTH && ks < LENGTH) v = W_bin[n * 150 + seg * LENGTH + ks];
      pb[idx] = (_Float16)v;
    }
    _Float16* pl = (_Float16*)sPackLeaf;
    for (int idx = tid; idx < 8 * 32 * 16; idx += 256) {
      int j  = idx & 15;
      int ln = (idx >> 4) & 31;
      int f  = idx >> 9;
      int nt = f & 3, kc = f >> 2;
      int n = nt * 16 + (ln & 15);
      int k = kc * 32 + ((ln >> 4) << 4) + j;
      float v = 0.f;
      if (n < LENGTH && k < ENC_DIM) v = W_leaf[n * ENC_DIM + k];
      pl[idx] = (_Float16)v;
    }
    for (int idx = tid; idx < 64; idx += 256) {
      sBiasBin[idx]  = (idx < LENGTH) ? b_bin[idx]  : 0.f;
      sBiasLeaf[idx] = (idx < LENGTH) ? b_leaf[idx] : 0.f;
    }
  }
  __syncthreads();

  const float biasBw = sBiasBin[ntw * 16 + n15];
  const float biasLw = sBiasLeaf[ntw * 16 + n15];

  // A-fragment assembly from a padded-64 LDS row triple (seg per k-chunk).
  auto loadA = [&](const _Float16* e0, const _Float16* e1, const _Float16* e2,
                   int ck) -> v16h {
    const _Float16* seg = (ck < 2) ? e0 : (ck < 4) ? e1 : e2;
    int hb = ((ck & 1) << 5) + laneHi * 16;
    v8h lo = *(const v8h*)(seg + hb);
    v8h hi = *(const v8h*)(seg + hb + 8);
    return __builtin_shufflevector(lo, hi, 0,1,2,3,4,5,6,7,8,9,10,11,12,13,14,15);
  };
  auto storeC = [&](_Float16* dst, const v8f& c, float bias) {
#pragma unroll
    for (int g = 0; g < 8; ++g) {
      float v = c[g] + bias;
      v = v > 0.f ? v : 0.f;
      dst[g * 64] = (_Float16)v;
    }
  };

  // ---------------- Stage 1: embeddings via WMMA (two-tile pipelined) ----------------
  for (int ch = 0; ch < 8; ++ch) {
    const bool isOp   = (ch >= 4);
    const int  chunk  = isOp ? (ch - 4) : ch;
    const int  maxRow = isOp ? (TTREES * 255 - 1) : (TTREES * 256 - 1);

    for (int idx = tid; idx < 128 * 16; idx += 256) {
      int r  = idx >> 4;
      int e4 = (idx & 15) << 2;
      int gl = chunk * 128 + r;
      if (gl > maxRow) gl = maxRow;
      int id;
      const float* tab;
      if (isOp) { int t = gl / 255; int i = gl - t * 255; id = op_ids[(b0 + t) * 255 + i]; tab = op_table; }
      else      { int t = gl >> 8;  int nd = gl & 255;    id = leaf_ids[(b0 + t) * 256 + nd]; tab = leaf_table; }
      const float* src = tab + id * ENC_DIM + e4;
      v4h p;
      p.x = (_Float16)((e4 + 0 < ENC_DIM) ? src[0] : 0.f);
      p.y = (_Float16)((e4 + 1 < ENC_DIM) ? src[1] : 0.f);
      p.z = (_Float16)((e4 + 2 < ENC_DIM) ? src[2] : 0.f);
      p.w = (_Float16)((e4 + 3 < ENC_DIM) ? src[3] : 0.f);
      *(v4h*)(&sB[r * 64 + e4]) = p;
    }
    __syncthreads();

    _Float16* base = (isOp ? sOp : sA) + (chunk * 128) * 64 + ntw * 16 + n15;
    for (int mt = (wave >> 2); mt < 8; mt += 4) {         // pair (mt, mt+2)
      const _Float16* a0 = &sB[(mt * 16 + n15) * 64];
      const _Float16* a1 = &sB[((mt + 2) * 16 + n15) * 64];
      v8f c0 = {}, c1 = {};
#pragma unroll
      for (int kc = 0; kc < 2; ++kc) {
        v16h f0 = loadA(a0, a0, a0, kc);
        v16h f1 = loadA(a1, a1, a1, kc);
        v16h b  = sPackLeaf[(kc * 4 + ntw) * 32 + lane];
        c0 = __builtin_amdgcn_wmma_f32_16x16x32_f16(false, f0, false, b, (short)0, c0, false, false);
        c1 = __builtin_amdgcn_wmma_f32_16x16x32_f16(false, f1, false, b, (short)0, c1, false, false);
      }
      storeC(base + (mt * 16 + laneHi * 8) * 64, c0, biasLw);
      storeC(base + ((mt + 2) * 16 + laneHi * 8) * 64, c1, biasLw);
    }
    __syncthreads();
  }

  // ---------------- Stage 2: tree contraction via WMMA (two-tile pipelined) ----------------
  int n_in = 256;
  int off  = 0;
  int sh   = 7;                  // log2(n_out) of first level
  _Float16* cur = sA;
  _Float16* nxt = sB;
  while (n_in > 1) {
    const int n_out    = n_in >> 1;
    const int rows_out = TTREES * n_out;
    const int Mtiles   = (rows_out + 15) >> 4;

    v16h bf[6];
#pragma unroll
    for (int kc = 0; kc < 6; ++kc) bf[kc] = sPackBin[(kc * 4 + ntw) * 32 + lane];

    auto tilePtrs = [&](int mt, const _Float16*& e0, const _Float16*& e1) {
      int row = mt * 16 + n15;
      if (row >= rows_out) row = rows_out - 1;
      int t = row >> sh;
      int j = row & (n_out - 1);
      e0 = cur + (t * n_in + 2 * j) * 64;
      e1 = sOp + (t * 255 + off + j) * 64;
    };
    _Float16* dbase = nxt + (laneHi * 8) * 64 + ntw * 16 + n15;

    for (int mt = (wave >> 2); mt < Mtiles; mt += 4) {    // pair (mt, mt+2)
      const bool two = (mt + 2) < Mtiles;                 // wave-uniform
      const _Float16 *p0, *q0, *p1, *q1;
      tilePtrs(mt, p0, q0);
      if (two) {
        tilePtrs(mt + 2, p1, q1);
        v8f c0 = {}, c1 = {};
#pragma unroll
        for (int ck = 0; ck < 6; ++ck) {
          v16h f0 = loadA(p0, q0, p0 + 64, ck);
          v16h f1 = loadA(p1, q1, p1 + 64, ck);
          c0 = __builtin_amdgcn_wmma_f32_16x16x32_f16(false, f0, false, bf[ck], (short)0, c0, false, false);
          c1 = __builtin_amdgcn_wmma_f32_16x16x32_f16(false, f1, false, bf[ck], (short)0, c1, false, false);
        }
        storeC(dbase + (mt * 16) * 64, c0, biasBw);
        storeC(dbase + ((mt + 2) * 16) * 64, c1, biasBw);
      } else {
        v8f c0 = {};
#pragma unroll
        for (int ck = 0; ck < 6; ++ck) {
          v16h f0 = loadA(p0, q0, p0 + 64, ck);
          c0 = __builtin_amdgcn_wmma_f32_16x16x32_f16(false, f0, false, bf[ck], (short)0, c0, false, false);
        }
        storeC(dbase + (mt * 16) * 64, c0, biasBw);
        if (n_out == 1) {                                 // final level: root in f32
#pragma unroll
          for (int g = 0; g < 8; ++g) {
            int r = mt * 16 + g + laneHi * 8;
            float v = c0[g] + biasBw;
            v = v > 0.f ? v : 0.f;
            if (r < TTREES) sRoot[r * 64 + ntw * 16 + n15] = v;
          }
        }
      }
    }
    __syncthreads();
    _Float16* tmp = cur; cur = nxt; nxt = tmp;
    off  += n_out;
    n_in  = n_out;
    --sh;
  }

  // ---------------- Stage 3: final 50 -> 55 layer (tiny, VALU) ----------------
  for (int idx = tid; idx < TTREES * ENC_DIM; idx += 256) {
    int t = idx / ENC_DIM;
    int d = idx - t * ENC_DIM;
    float s = b_final[d];
    const float* wr = &W_final[d * LENGTH];
    const float* rr = &sRoot[t * 64];
#pragma unroll
    for (int k = 0; k < LENGTH; ++k) s += rr[k] * wr[k];
    out[(b0 + t) * ENC_DIM + d] = s;
  }
}

extern "C" void kernel_launch(void* const* d_in, const int* in_sizes, int n_in,
                              void* d_out, int out_size, void* d_ws, size_t ws_size,
                              hipStream_t stream) {
  const float* leaf_table = (const float*)d_in[0];
  const float* op_table   = (const float*)d_in[1];
  const float* W_leaf     = (const float*)d_in[2];
  const float* b_leaf     = (const float*)d_in[3];
  const float* W_bin      = (const float*)d_in[4];
  const float* b_bin      = (const float*)d_in[5];
  const float* W_final    = (const float*)d_in[6];
  const float* b_final    = (const float*)d_in[7];
  const int*   leaf_ids   = (const int*)d_in[8];
  const int*   op_ids     = (const int*)d_in[9];
  float* outp = (float*)d_out;

  const int B = 4096;
  dim3 grid(B / TTREES), block(256);
  formula_net_kernel<<<grid, block, 0, stream>>>(
      leaf_table, op_table, W_leaf, b_leaf, W_bin, b_bin,
      W_final, b_final, leaf_ids, op_ids, outp);
}
